// Mutation_GraphSAGE_12232066859618
// MI455X (gfx1250) — matrix operand order repaired and checked by reading, not compiled, not run on hardware
//
#include <hip/hip_runtime.h>
#include <stdint.h>

typedef __attribute__((ext_vector_type(16))) _Float16 v16h;
typedef __attribute__((ext_vector_type(8)))  float    v8f;

#define NNODES 50000
#define NEDGES 800000
#define FDIM   128     // input / hidden feature width
#define KDIM   256     // concat(mean, self) contraction length

// ---------------------------------------------------------------- degree count
__global__ void k_count(const long long* __restrict__ dst, float* __restrict__ cnt, int E) {
    int i = blockIdx.x * blockDim.x + threadIdx.x;
    if (i < E) atomicAdd(&cnt[(int)dst[i]], 1.0f);
}

__global__ void k_inv(float* __restrict__ cnt, int n) {
    int i = blockIdx.x * blockDim.x + threadIdx.x;
    if (i < n) cnt[i] = 1.0f / fmaxf(cnt[i], 1.0f);
}

// ------------------------------------------------- edge scatter-add (mean num.)
// one wave per edge, 4 consecutive floats per lane (float4 coalesced load)
__global__ void k_agg(const long long* __restrict__ src, const long long* __restrict__ dst,
                      const float* __restrict__ x, float* __restrict__ agg, int E) {
    int t = blockIdx.x * blockDim.x + threadIdx.x;
    int e = t >> 5;
    if (e >= E) return;
    int l = t & 31;
    int s = (int)src[e];
    int d = (int)dst[e];
    const float4 v = *(const float4*)(x + (size_t)s * FDIM + l * 4);
    float* p = agg + (size_t)d * FDIM + l * 4;
    atomicAdd(p + 0, v.x);
    atomicAdd(p + 1, v.y);
    atomicAdd(p + 2, v.z);
    atomicAdd(p + 3, v.w);
}

// --------------------------- fuse mean divide + concat(self) + f32->f16 convert
// A[i, k] = k<128 ? agg[i,k] * inv[i] : xin[i, k-128]
__global__ void k_prepA(const float* __restrict__ agg, const float* __restrict__ inv,
                        const float* __restrict__ xin, _Float16* __restrict__ A) {
    int t = blockIdx.x * blockDim.x + threadIdx.x;
    if (t >= NNODES * KDIM) return;
    int i = t >> 8;
    int k = t & 255;
    float v = (k < FDIM) ? agg[(size_t)i * FDIM + k] * inv[i]
                         : xin[(size_t)i * FDIM + (k - FDIM)];
    A[t] = (_Float16)v;
}

// pack B[j, 0:256] = [ Wl[j, 0:128] | Wr[j, 0:128] ] as f16
__global__ void k_prepB(const float* __restrict__ Wl, const float* __restrict__ Wr,
                        _Float16* __restrict__ B, int rows) {
    int t = blockIdx.x * blockDim.x + threadIdx.x;
    if (t >= rows * KDIM) return;
    int j = t >> 8;
    int k = t & 255;
    float v = (k < FDIM) ? Wl[(size_t)j * FDIM + k] : Wr[(size_t)j * FDIM + (k - FDIM)];
    B[t] = (_Float16)v;
}

// ------------------------------------------------------------- WMMA tile GEMM
// out[M, Hout] = A[M,256] @ B[Hout,256]^T (+bias, optional relu)
// one wave per 16x16 output tile; K looped in 8 steps of v_wmma_f32_16x16x32_f16.
// Fragment layouts per CDNA5 ISA 7.12.2 (wave32):
//   A 16x32 f16: lane (l&15)=row M; half=l>>4; elems 0..7 -> K = 8h+0..7,
//                elems 8..15 -> K = 16+8h+0..7       (two contiguous uint4 loads)
//   B 32x16 f16: lane (l&15)=col N; same K mapping as A (two contiguous uint4 loads)
//   C/D 16x16 f32: lane (l&15)=col N; VGPR r -> row M = r + 8h
__global__ void k_gemm(const _Float16* __restrict__ A, const _Float16* __restrict__ B,
                       const float* __restrict__ bias, float* __restrict__ out,
                       int numColTiles, int totalTiles, int Hout, int doRelu) {
    int wave = (blockIdx.x * blockDim.x + threadIdx.x) >> 5;
    int lane = threadIdx.x & 31;
    if (wave >= totalTiles) return;           // uniform per wave: EXEC all-1s inside

    int tr   = wave / numColTiles;
    int tc   = wave % numColTiles;
    int half = lane >> 4;
    int l15  = lane & 15;

    const _Float16* Arow = A + (size_t)(tr * 16 + l15) * KDIM;
    const _Float16* Brow = B + (size_t)(tc * 16 + l15) * KDIM;

    v8f c = {};
#pragma unroll
    for (int kk = 0; kk < 8; ++kk) {
        const int k0 = kk * 32;
        v16h a, b;
        *(uint4*)&a         = *(const uint4*)(Arow + k0 + half * 8);
        *((uint4*)&a + 1)   = *(const uint4*)(Arow + k0 + 16 + half * 8);
        *(uint4*)&b         = *(const uint4*)(Brow + k0 + half * 8);
        *((uint4*)&b + 1)   = *(const uint4*)(Brow + k0 + 16 + half * 8);
        c = __builtin_amdgcn_wmma_f32_16x16x32_f16(
                /*neg_a=*/false, a, /*neg_b=*/false, b,
                /*c_mod=*/(short)0, c, /*reuse_a=*/false, /*reuse_b=*/false);
    }

    const int col = tc * 16 + l15;
    const float bv = bias[col];
#pragma unroll
    for (int r = 0; r < 8; ++r) {
        int m = tr * 16 + half * 8 + r;
        float v = c[r] + bv;
        if (doRelu) v = fmaxf(v, 0.0f);
        out[(size_t)m * Hout + col] = v;
    }
}

// -------------------------------------------- in-place log_softmax over 64 cols
__global__ void k_logsoftmax(float* __restrict__ out, int n) {
    int wave = (blockIdx.x * blockDim.x + threadIdx.x) >> 5;
    if (wave >= n) return;
    int lane = threadIdx.x & 31;
    float* row = out + (size_t)wave * 64;
    float v0 = row[lane];
    float v1 = row[lane + 32];
    float m = fmaxf(v0, v1);
#pragma unroll
    for (int off = 16; off > 0; off >>= 1) m = fmaxf(m, __shfl_xor(m, off, 32));
    float s = __expf(v0 - m) + __expf(v1 - m);
#pragma unroll
    for (int off = 16; off > 0; off >>= 1) s += __shfl_xor(s, off, 32);
    float lse = m + __logf(s);
    row[lane]      = v0 - lse;
    row[lane + 32] = v1 - lse;
}

static inline size_t align256(size_t v) { return (v + 255) & ~(size_t)255; }

extern "C" void kernel_launch(void* const* d_in, const int* in_sizes, int n_in,
                              void* d_out, int out_size, void* d_ws, size_t ws_size,
                              hipStream_t stream) {
    (void)in_sizes; (void)n_in; (void)out_size; (void)ws_size;

    const float*     x    = (const float*)d_in[0];
    const long long* ei   = (const long long*)d_in[1];   // int64 [2, E] flat
    const float*     W1l  = (const float*)d_in[2];
    const float*     b1   = (const float*)d_in[3];
    const float*     W1r  = (const float*)d_in[4];
    const float*     W2l  = (const float*)d_in[5];
    const float*     b2   = (const float*)d_in[6];
    const float*     W2r  = (const float*)d_in[7];
    float*           outp = (float*)d_out;

    const long long* src = ei;
    const long long* dst = ei + NEDGES;

    // workspace carve-up
    char* w = (char*)d_ws;
    size_t o = 0;
    float*    cnt = (float*)(w + o);    o = align256(o + (size_t)NNODES * 4);
    float*    agg = (float*)(w + o);    o = align256(o + (size_t)NNODES * FDIM * 4);
    _Float16* Abf = (_Float16*)(w + o); o = align256(o + (size_t)NNODES * KDIM * 2);
    _Float16* Bbf = (_Float16*)(w + o); o = align256(o + (size_t)FDIM * KDIM * 2);
    float*    h   = (float*)(w + o);    o = align256(o + (size_t)NNODES * FDIM * 4);

    const int TPB = 256;

    // ---- degree (shared by both layers)
    hipMemsetAsync(cnt, 0, (size_t)NNODES * 4, stream);
    k_count<<<(NEDGES + TPB - 1) / TPB, TPB, 0, stream>>>(dst, cnt, NEDGES);
    k_inv<<<(NNODES + TPB - 1) / TPB, TPB, 0, stream>>>(cnt, NNODES);

    // ---- layer 1: h = relu([mean(x)|x] @ [W1l|W1r]^T + b1)
    hipMemsetAsync(agg, 0, (size_t)NNODES * FDIM * 4, stream);
    {
        long long tot = (long long)NEDGES * 32;
        k_agg<<<(unsigned)((tot + TPB - 1) / TPB), TPB, 0, stream>>>(src, dst, x, agg, NEDGES);
    }
    k_prepB<<<(FDIM * KDIM + TPB - 1) / TPB, TPB, 0, stream>>>(W1l, W1r, Bbf, FDIM);
    k_prepA<<<(NNODES * KDIM + TPB - 1) / TPB, TPB, 0, stream>>>(agg, cnt, x, Abf);
    {
        const int colTiles = FDIM / 16;                 // 8
        const int tiles    = (NNODES / 16) * colTiles;  // 25000
        const int blocks   = (tiles * 32 + TPB - 1) / TPB;
        k_gemm<<<blocks, TPB, 0, stream>>>(Abf, Bbf, b1, h, colTiles, tiles, FDIM, 1);
    }

    // ---- layer 2: out = [mean(h)|h] @ [W2l|W2r]^T + b2
    hipMemsetAsync(agg, 0, (size_t)NNODES * FDIM * 4, stream);
    {
        long long tot = (long long)NEDGES * 32;
        k_agg<<<(unsigned)((tot + TPB - 1) / TPB), TPB, 0, stream>>>(src, dst, h, agg, NEDGES);
    }
    k_prepB<<<(64 * KDIM + TPB - 1) / TPB, TPB, 0, stream>>>(W2l, W2r, Bbf, 64);
    k_prepA<<<(NNODES * KDIM + TPB - 1) / TPB, TPB, 0, stream>>>(agg, cnt, h, Abf);
    {
        const int colTiles = 64 / 16;                   // 4
        const int tiles    = (NNODES / 16) * colTiles;  // 12500
        const int blocks   = (tiles * 32 + TPB - 1) / TPB;
        k_gemm<<<blocks, TPB, 0, stream>>>(Abf, Bbf, b2, outp, colTiles, tiles, 64, 0);
    }

    // ---- log_softmax rows of 64, in place
    {
        const int blocks = (NNODES * 32 + TPB - 1) / TPB;
        k_logsoftmax<<<blocks, TPB, 0, stream>>>(outp, NNODES);
    }
}
